// DecoderGru_8847632630530
// MI455X (gfx1250) — matrix operand order repaired and checked by reading, not compiled, not run on hardware
//
#include <hip/hip_runtime.h>
#include <hip/hip_bf16.h>

// ---------------------------------------------------------------------------
// Types for CDNA5 WMMA (wave32, gfx1250)
// ---------------------------------------------------------------------------
typedef __attribute__((ext_vector_type(16))) __bf16 v16bf;
typedef __attribute__((ext_vector_type(2)))  __bf16 v2bf;
typedef __attribute__((ext_vector_type(8)))  float  v8f;
typedef __attribute__((ext_vector_type(4)))  float  vf4;
typedef __attribute__((ext_vector_type(4)))  unsigned int v4u;

union Frag {
    v4u   q[2];   // two 16-byte halves (8 bf16 each)
    v16bf v;
};

__device__ __forceinline__ unsigned short f32_to_bf16_bits(float f) {
    union { float f; unsigned int u; } x; x.f = f;
    unsigned int r = x.u + 0x7FFFu + ((x.u >> 16) & 1u);   // round-to-nearest-even
    return (unsigned short)(r >> 16);
}

#if __has_builtin(__builtin_amdgcn_cvt_pk_bf16_f32)
__device__ __forceinline__ unsigned int pack2_bf16(float a, float b) {
    union { v2bf v; unsigned int u; } c;
    c.v = __builtin_amdgcn_cvt_pk_bf16_f32(a, b);
    return c.u;
}
#else
__device__ __forceinline__ unsigned int pack2_bf16(float a, float b) {
    union { float f; unsigned int u; } xa, xb; xa.f = a; xb.f = b;
    unsigned int ra = xa.u + 0x7FFFu + ((xa.u >> 16) & 1u);
    unsigned int rb = xb.u + 0x7FFFu + ((xb.u >> 16) & 1u);
    return (ra >> 16) | (rb & 0xFFFF0000u);
}
#endif

// ---------------------------------------------------------------------------
// Constants
// ---------------------------------------------------------------------------
#define BATCH 128
#define HID   1024
#define EMB   300
#define EMBP  320          // padded to 10 K-tiles of 32
#define G3H   3072         // 3*HID
#define VOCAB 50257

// ---------------------------------------------------------------------------
// Kernel 1: embedding gather + f32 -> bf16 casts of the A panels
// ---------------------------------------------------------------------------
__global__ __launch_bounds__(256)
void embed_cast_kernel(const int* __restrict__ ids,
                       const float* __restrict__ hprev,
                       const float* __restrict__ emb,
                       unsigned short* __restrict__ Ax,
                       unsigned short* __restrict__ Ah)
{
    const int b  = blockIdx.x;
    const long id = (long)ids[b];
    for (int k = threadIdx.x; k < EMBP; k += blockDim.x) {
        float v = (k < EMB) ? emb[id * EMB + k] : 0.0f;
        Ax[b * EMBP + k] = f32_to_bf16_bits(v);
    }
    for (int k = threadIdx.x * 2; k < HID; k += blockDim.x * 2) {
        ((unsigned int*)Ah)[(b * HID + k) >> 1] =
            pack2_bf16(hprev[b * HID + k], hprev[b * HID + k + 1]);
    }
}

// ---------------------------------------------------------------------------
// Kernel 2: bf16-WMMA GEMM  out(128 x nrows) = A(128 x K) * W(nrows x K)^T + bias
//   Double-buffered K pipeline: iteration kt consumes LDS buffer kt&1 while
//   issuing tile kt+1's loads (async global->LDS for A, b128+pack for B) into
//   the other buffer, overlapping memory latency with the 4 WMMAs.
// ---------------------------------------------------------------------------
#define LDSTR 40
#define ASZ   (128 * LDSTR)
#define BSZ   (64  * LDSTR)

__global__ __launch_bounds__(256)
void gemm_bf16_kernel(const unsigned short* __restrict__ A, int lda,
                      const float* __restrict__ W, int ldw, int nrows,
                      int kreal, int ktiles,
                      const float* __restrict__ bias,
                      float* __restrict__ out, int ldo)
{
    __shared__ unsigned short As[2 * ASZ];
    __shared__ unsigned short Bs[2 * BSZ];

    const int tid  = threadIdx.x;
    const int lane = tid & 31;
    const int wave = tid >> 5;
    const int wm   = (wave & 3) * 32;      // wave M offset within block tile
    const int wn   = (wave >> 2) * 32;     // wave N offset within block tile
    const int nblk = blockIdx.x * 64;      // block N offset (rows of W)

    v8f acc[2][2] = {};                    // 2x2 wmma tiles per wave

    // fragment gather coordinates (per documented CDNA5 VGPR layouts)
    const int am   = lane & 15;            // A: M within 16-tile
    const int aklo = (lane >> 4) * 8;      // A: K low offset per half-wave
    const int bn   = lane & 15;            // B: N within 16-tile
    const int bkb  = (lane >> 4) * 8;      // B: K low offset per half-wave

    // per-thread staging coordinates
    const int sam = tid >> 1;              // A stage: row 0..127
    const int sak = (tid & 1) * 16;        // A stage: 16-short half
    const int sbn = tid >> 2;              // B stage: row 0..63
    const int sbk = (tid & 3) * 8;         // B stage: 8-float chunk

    const bool nfull = (nblk + 64) <= nrows;

    // stage tile kt into LDS buffer `buf`
    auto stage = [&](int kt, int buf) {
        const int k0 = kt * 32;
        unsigned short* Asb = As + buf * ASZ;
        unsigned short* Bsb = Bs + buf * BSZ;

        // A tile (128 x 32 bf16): async global -> LDS (32 B per thread)
        {
            unsigned long long g0 =
                (unsigned long long)(A + sam * lda + k0 + sak);
            unsigned int l0 =
                (unsigned int)(unsigned long long)(Asb + sam * LDSTR + sak);
            asm volatile(
                "global_load_async_to_lds_b128 %0, %1, off\n\t"
                "global_load_async_to_lds_b128 %0, %1, off offset:16"
                :: "v"(l0), "v"(g0) : "memory");
        }
        // B tile (64 x 32): f32 -> bf16
        if (nfull && (k0 + 32) <= kreal) {
            const float* src = W + (long)(nblk + sbn) * ldw + k0 + sbk;
            vf4 f0 = *(const vf4*)(src);
            vf4 f1 = *(const vf4*)(src + 4);
            v4u p = { pack2_bf16(f0[0], f0[1]), pack2_bf16(f0[2], f0[3]),
                      pack2_bf16(f1[0], f1[1]), pack2_bf16(f1[2], f1[3]) };
            *(v4u*)(Bsb + sbn * LDSTR + sbk) = p;
        } else {
            const int grow = nblk + sbn;
            unsigned short* dst = Bsb + sbn * LDSTR + sbk;
            #pragma unroll
            for (int i = 0; i < 8; ++i) {
                const int k = k0 + sbk + i;
                float v = 0.0f;
                if (grow < nrows && k < kreal)
                    v = W[(long)grow * ldw + k];
                dst[i] = f32_to_bf16_bits(v);
            }
        }
    };

    stage(0, 0);

    for (int kt = 0; kt < ktiles; ++kt) {
        const int cur = kt & 1;

        // tile kt is the only outstanding async batch for this wave
        asm volatile("s_wait_asynccnt 0x0" ::: "memory");
        __syncthreads();   // buf[cur] fully written; prior reads of buf[1-cur] drained

        const unsigned short* Asb = As + cur * ASZ;
        const unsigned short* Bsb = Bs + cur * BSZ;

        // ---- build fragments for tile kt ----
        Frag a[2], b[2];
        #pragma unroll
        for (int mi = 0; mi < 2; ++mi) {
            const unsigned short* p = Asb + (wm + mi * 16 + am) * LDSTR;
            a[mi].q[0] = *(const v4u*)(p + aklo);        // K = klo .. klo+7
            a[mi].q[1] = *(const v4u*)(p + aklo + 16);   // K = klo+16 .. klo+23
        }
        #pragma unroll
        for (int ni = 0; ni < 2; ++ni) {
            const unsigned short* p = Bsb + (wn + ni * 16 + bn) * LDSTR + bkb;
            b[ni].q[0] = *(const v4u*)(p);               // K = kb .. kb+7
            b[ni].q[1] = *(const v4u*)(p + 16);          // K = kb+16 .. kb+23
        }

        // ---- issue next tile's loads into the other buffer (overlaps WMMA) ----
        if (kt + 1 < ktiles)
            stage(kt + 1, cur ^ 1);

        // ---- 4 WMMAs per wave ----
        #pragma unroll
        for (int mi = 0; mi < 2; ++mi)
            #pragma unroll
            for (int ni = 0; ni < 2; ++ni)
                acc[mi][ni] = __builtin_amdgcn_wmma_f32_16x16x32_bf16(
                    false, a[mi].v, false, b[ni].v,
                    (short)0, acc[mi][ni], false, false);
    }

    // ---- epilogue: scatter C tiles (f32 16x16 layout) + bias ----
    const int cm = (lane < 16) ? 0 : 8;
    const int cn = lane & 15;
    #pragma unroll
    for (int mi = 0; mi < 2; ++mi) {
        #pragma unroll
        for (int ni = 0; ni < 2; ++ni) {
            const int n = nblk + wn + ni * 16 + cn;
            if (n < nrows) {
                const float bb = bias ? bias[n] : 0.0f;
                #pragma unroll
                for (int r = 0; r < 8; ++r) {
                    const int m = wm + mi * 16 + r + cm;   // batch row
                    out[(long)m * ldo + n] = acc[mi][ni][r] + bb;
                }
            }
        }
    }
}

// ---------------------------------------------------------------------------
// Kernel 3: fused GRU gates -> h_new (f32 to d_out tail, bf16 to ws panel)
// ---------------------------------------------------------------------------
__global__ __launch_bounds__(256)
void gru_gate_kernel(const float* __restrict__ gi,
                     const float* __restrict__ gh,
                     const float* __restrict__ hprev,
                     float* __restrict__ hnew,
                     unsigned short* __restrict__ Ahn)
{
    const int idx = blockIdx.x * blockDim.x + threadIdx.x;  // 0 .. 128*1024-1
    const int b = idx >> 10;
    const int j = idx & 1023;
    const long base = (long)b * G3H;

    const float ir = gi[base + j];
    const float iz = gi[base + HID + j];
    const float in = gi[base + 2 * HID + j];
    const float hr = gh[base + j];
    const float hz = gh[base + HID + j];
    const float hn = gh[base + 2 * HID + j];

    const float r = 1.0f / (1.0f + __expf(-(ir + hr)));
    const float z = 1.0f / (1.0f + __expf(-(iz + hz)));
    const float n = tanhf(in + r * hn);
    const float h = hprev[idx];
    const float hv = (1.0f - z) * n + z * h;

    hnew[idx] = hv;
    Ahn[idx]  = f32_to_bf16_bits(hv);
}

// ---------------------------------------------------------------------------
// Kernel 4: in-place log_softmax per batch row (online logsumexp, 2 passes)
//   1024 threads/block (32 waves) so the 77 MB of logits traffic has enough
//   lanes in flight to approach HBM peak.
// ---------------------------------------------------------------------------
#define SMT 1024

__global__ __launch_bounds__(SMT)
void logsoftmax_kernel(float* __restrict__ logits, int V)
{
    const int b = blockIdx.x;
    float* row = logits + (long)b * V;

    float mx = -INFINITY, s = 0.0f;
    for (int v = threadIdx.x; v < V; v += SMT) {
        const float x = row[v];
        if (x > mx) { s *= __expf(mx - x); mx = x; }
        s += __expf(x - mx);
    }

    __shared__ float smx[SMT], ssm[SMT];
    smx[threadIdx.x] = mx;
    ssm[threadIdx.x] = s;
    __syncthreads();
    for (int off = SMT / 2; off > 0; off >>= 1) {
        if (threadIdx.x < off) {
            const float m1 = smx[threadIdx.x], m2 = smx[threadIdx.x + off];
            const float s1 = ssm[threadIdx.x], s2 = ssm[threadIdx.x + off];
            const float m  = fmaxf(m1, m2);
            smx[threadIdx.x] = m;
            ssm[threadIdx.x] = s1 * __expf(m1 - m) + s2 * __expf(m2 - m);
        }
        __syncthreads();
    }
    const float M = smx[0];
    const float L = __logf(ssm[0]);

    for (int v = threadIdx.x; v < V; v += SMT)
        row[v] = row[v] - M - L;
}

// ---------------------------------------------------------------------------
// Host launch
// ---------------------------------------------------------------------------
extern "C" void kernel_launch(void* const* d_in, const int* in_sizes, int n_in,
                              void* d_out, int out_size, void* d_ws, size_t ws_size,
                              hipStream_t stream) {
    const int*   ids   = (const int*)  d_in[0];
    const float* hprev = (const float*)d_in[1];
    // d_in[2] = encoder_output (unused by reference)
    const float* emb   = (const float*)d_in[3];
    const float* w_ih  = (const float*)d_in[4];
    const float* w_hh  = (const float*)d_in[5];
    const float* b_ih  = (const float*)d_in[6];
    const float* b_hh  = (const float*)d_in[7];
    const float* W_out = (const float*)d_in[8];
    const float* b_out = (const float*)d_in[9];

    float* out_logits = (float*)d_out;                          // (128 x 50257)
    float* out_hnew   = (float*)d_out + (long)BATCH * VOCAB;    // (128 x 1024)

    // workspace layout (bytes)
    char* ws = (char*)d_ws;
    unsigned short* Ax  = (unsigned short*)(ws);                       // 128*320 bf16
    unsigned short* Ah  = (unsigned short*)(ws + 128 * 1024);          // 128*1024 bf16
    unsigned short* Ahn = (unsigned short*)(ws + 128 * 1024 + 256 * 1024);
    float* gi = (float*)(ws + 128 * 1024 + 512 * 1024);                // 128*3072 f32
    float* gh = (float*)(ws + 128 * 1024 + 512 * 1024 + 1536 * 1024);  // 128*3072 f32

    // 1) embedding gather + bf16 casts
    embed_cast_kernel<<<BATCH, 256, 0, stream>>>(ids, hprev, emb, Ax, Ah);

    // 2) input gates:  gi = x * w_ih^T + b_ih   (K=300 padded to 320)
    gemm_bf16_kernel<<<G3H / 64, 256, 0, stream>>>(
        Ax, EMBP, w_ih, EMB, G3H, EMB, EMBP / 32, b_ih, gi, G3H);

    // 3) hidden gates: gh = h * w_hh^T + b_hh
    gemm_bf16_kernel<<<G3H / 64, 256, 0, stream>>>(
        Ah, HID, w_hh, HID, G3H, HID, HID / 32, b_hh, gh, G3H);

    // 4) GRU nonlinearity -> h_new (output #2) + bf16 panel
    gru_gate_kernel<<<(BATCH * HID) / 256, 256, 0, stream>>>(
        gi, gh, hprev, out_hnew, Ahn);

    // 5) output projection: logits = h_new * W_out^T + b_out   (dominant GEMM)
    gemm_bf16_kernel<<<(VOCAB + 63) / 64, 256, 0, stream>>>(
        Ahn, HID, W_out, HID, VOCAB, HID, HID / 32, b_out, out_logits, VOCAB);

    // 6) in-place log_softmax over vocab per batch row
    logsoftmax_kernel<<<BATCH, SMT, 0, stream>>>(out_logits, VOCAB);
}
